// GraphTransformer_5592047419592
// MI455X (gfx1250) — matrix operand order repaired and checked by reading, not compiled, not run on hardware
//
#include <hip/hip_runtime.h>
#include <hip/hip_bf16.h>
#include <math.h>

// ---------------------------------------------------------------------------
// MI455X / gfx1250 implementation. wave32. All matrix math on
// v_wmma_f32_16x16x32_f16 (f16 inputs, f32 accumulate). Weight tiles staged
// through LDS (f16, transposed) and shared by all 4 waves of a block;
// A-fragments loaded as float4; next K-chunk prefetched (global_prefetch_b8).
// ---------------------------------------------------------------------------

typedef __attribute__((ext_vector_type(16))) _Float16 v16h;
typedef __attribute__((ext_vector_type(8)))  _Float16 v8h;
typedef __attribute__((ext_vector_type(4)))  _Float16 v4h;
typedef __attribute__((ext_vector_type(8)))  float    v8f;
typedef __attribute__((ext_vector_type(4)))  float    v4f;

#define N_NODES 4096
#define N_EDGES 65536
#define HIDC    256           // HID
#define MHA_D   192
#define MHA_HD  48

__device__ inline v8f wmma_f16(v16h a, v16h b, v8f c) {
  return __builtin_amdgcn_wmma_f32_16x16x32_f16(false, a, false, b, (short)0, c,
                                                false, false);
}

// ordered-key float atomic max via uint
__device__ inline unsigned fkey(float f) {
  unsigned u = __float_as_uint(f);
  return (u & 0x80000000u) ? ~u : (u | 0x80000000u);
}
__device__ inline float funkey(unsigned u) {
  return (u & 0x80000000u) ? __uint_as_float(u & 0x7fffffffu)
                           : __uint_as_float(~u);
}

// ---------------------------------------------------------------------------
// Generic WMMA GEMM: C[M,N] = A[M,K] @ W[K,N] + bias.
// A rows optionally gathered through `arows`. K % 32 == 0, N % 16 == 0.
// Block = 128 threads (4 waves): all waves share one 16-col tile (blockIdx.y),
// waves stacked along M. Per K-chunk the 32x16 W tile is staged to LDS as f16
// transposed [n][k] so B-fragments are two contiguous ds_load_b128 per lane.
// Fragment layouts per CDNA5 ISA 7.12.2 (wave32):
//   A 16x32 f16 : lane l -> M = l&15;  elems 0..7 = K (l>>4)*8 + e,
//                                        elems 8..15 = K 16 + (l>>4)*8 + e
//   B 32x16 f16 : lane l -> N = l&15;  elems 0..7 = K (l>>4)*8 + e, 8..15 = K 16 + (l>>4)*8 + e
//   C/D 16x16 f32: lane l, elem r -> row r + 8*(l>>4), col l&15
// ---------------------------------------------------------------------------
__global__ void k_gemm_wmma(const float* __restrict__ A, long lda, long acol,
                            const int* __restrict__ arows,
                            const float* __restrict__ W, int ldw,
                            const float* __restrict__ bias,
                            float* __restrict__ C, int ldc, int ccol,
                            int M, int N, int K)
{
  __shared__ _Float16 sW[16 * 32];             // [n][k], stride 32 halfs

  const int wave = threadIdx.x >> 5;
  const int lane = threadIdx.x & 31;
  const int row0 = (blockIdx.x * 4 + wave) * 16;
  const int col0 = blockIdx.y * 16;
  // clamp out-of-range waves to a valid tile so every wave reaches all
  // barriers and participates in staging; only the store is guarded.
  const int r0 = (row0 < M) ? row0 : (M - 16);

  const int m  = lane & 15;
  const int hi = lane >> 4;
  long ar = r0 + m;
  if (arows) ar = arows[ar];
  const float* ap = A + ar * lda + acol;

  const int sn = threadIdx.x & 15;             // staging: col within tile
  const int kq = threadIdx.x >> 4;             // staging: k quad 0..7

  v8f acc = {};
  for (int k0 = 0; k0 < K; k0 += 32) {
    __syncthreads();                           // previous chunk fully consumed
    {                                          // stage W[k0..k0+31][col0..+15]
      const float* wsrc = W + (long)(k0 + kq * 4) * ldw + col0 + sn;
      v4h wv;
#pragma unroll
      for (int i = 0; i < 4; ++i) wv[i] = (_Float16)wsrc[(long)i * ldw];
      *(v4h*)&sW[sn * 32 + kq * 4] = wv;
    }
    if (k0 + 32 < K) {                         // prefetch next chunk
      __builtin_prefetch(ap + k0 + 32, 0, 3);
      __builtin_prefetch(W + (long)(k0 + 32 + kq * 4) * ldw + col0 + sn, 0, 3);
    }
    // A fragment: four contiguous float4 loads, convert to f16
    const v4f a0 = *(const v4f*)(ap + k0 + hi * 8);
    const v4f a1 = *(const v4f*)(ap + k0 + hi * 8 + 4);
    const v4f a2 = *(const v4f*)(ap + k0 + 16 + hi * 8);
    const v4f a3 = *(const v4f*)(ap + k0 + 16 + hi * 8 + 4);
    v16h a;
#pragma unroll
    for (int e = 0; e < 4; ++e) {
      a[e]      = (_Float16)a0[e];
      a[e + 4]  = (_Float16)a1[e];
      a[e + 8]  = (_Float16)a2[e];
      a[e + 12] = (_Float16)a3[e];
    }
    __syncthreads();                           // staging visible
    const _Float16* bp = &sW[m * 32 + hi * 8];
    const v8h b0 = *(const v8h*)bp;            // ds_load_b128
    const v8h b1 = *(const v8h*)(bp + 16);     // ds_load_b128
    v16h b;
#pragma unroll
    for (int e = 0; e < 8; ++e) { b[e] = b0[e]; b[e + 8] = b1[e]; }
    acc = wmma_f16(a, b, acc);
  }

  if (row0 < M) {
    const int col = col0 + m;
    const float bv = bias ? bias[col] : 0.f;
#pragma unroll
    for (int r = 0; r < 8; ++r) {
      const int row = r0 + r + 8 * hi;
      C[(long)row * ldc + ccol + col] = acc[r] + bv;
    }
  }
}

// ---------------------------------------------------------------------------
// Dense MHA over nodes (flash-attention style, one wave per (head, 16-q tile)).
// q/k/v/o are [4096,192] f32, head h covers dims h*48..h*48+47.
// Score WMMA: head dim 48 padded to 64 (two K=32 steps, tail zeroed).
// ---------------------------------------------------------------------------
__global__ void k_mha_attn(const float* __restrict__ Q,
                           const float* __restrict__ Km,
                           const float* __restrict__ V,
                           float* __restrict__ O)
{
  const int lane = threadIdx.x;
  const int h  = blockIdx.y;
  const int q0 = blockIdx.x * 16;
  const int m  = lane & 15;
  const int hi = lane >> 4;

  __shared__ float    sS[16 * 32];
  __shared__ _Float16 sP[16 * 32];
  __shared__ float sAlpha[16], sSum[16], sMax[16];

  const float* qp = Q + (long)(q0 + m) * MHA_D + h * MHA_HD;
  v16h aq0, aq1;
  {
    const v4f q0v = *(const v4f*)(qp + hi * 8);
    const v4f q1v = *(const v4f*)(qp + hi * 8 + 4);
    const v4f q2v = *(const v4f*)(qp + 16 + hi * 8);
    const v4f q3v = *(const v4f*)(qp + 16 + hi * 8 + 4);
    const v4f q4v = *(const v4f*)(qp + 32 + hi * 8);
    const v4f q5v = *(const v4f*)(qp + 32 + hi * 8 + 4);
#pragma unroll
    for (int e = 0; e < 4; ++e) {
      aq0[e]      = (_Float16)q0v[e];
      aq0[e + 4]  = (_Float16)q1v[e];
      aq0[e + 8]  = (_Float16)q2v[e];
      aq0[e + 12] = (_Float16)q3v[e];
      aq1[e]      = (_Float16)q4v[e];       // dims 32..47
      aq1[e + 4]  = (_Float16)q5v[e];
      aq1[e + 8]  = (_Float16)0.f;          // pad 48..63
      aq1[e + 12] = (_Float16)0.f;
    }
  }
  if (lane < 16) { sMax[lane] = -3.0e38f; sSum[lane] = 0.f; }
  __syncthreads();

  v8f o0 = {}, o1 = {}, o2 = {};
  const float scale = 0.14433756729740643f;       // 1/sqrt(48)

  for (int kt = 0; kt < N_NODES; kt += 32) {
#pragma unroll
    for (int sub = 0; sub < 2; ++sub) {
      const float* kp = Km + (long)(kt + sub * 16 + m) * MHA_D + h * MHA_HD;
      // b0: dims hi*8..hi*8+7, 16+hi*8..16+hi*8+7; b1: dims 32+hi*8..32+hi*8+7
      // in elems 0..7, elems 8..15 zero (pad 48..63).
      const v4f k0v = *(const v4f*)(kp + hi * 8);
      const v4f k1v = *(const v4f*)(kp + hi * 8 + 4);
      const v4f k2v = *(const v4f*)(kp + 16 + hi * 8);
      const v4f k3v = *(const v4f*)(kp + 16 + hi * 8 + 4);
      const v4f p0v = *(const v4f*)(kp + 32 + hi * 8);
      const v4f p1v = *(const v4f*)(kp + 36 + hi * 8);
      const v4f p2v = *(const v4f*)(kp + 40);
      const v4f p3v = *(const v4f*)(kp + 44);
      v16h b0, b1;
#pragma unroll
      for (int e = 0; e < 4; ++e) {
        b0[e]      = (_Float16)k0v[e];
        b0[e + 4]  = (_Float16)k1v[e];
        b0[e + 8]  = (_Float16)k2v[e];
        b0[e + 12] = (_Float16)k3v[e];
        b1[e]      = (_Float16)p0v[e];
        b1[e + 4]  = (_Float16)p1v[e];
        b1[e + 8]  = (_Float16)0.f;
        b1[e + 12] = (_Float16)0.f;
      }
      v8f s = {};
      s = wmma_f16(aq0, b0, s);
      s = wmma_f16(aq1, b1, s);
#pragma unroll
      for (int r = 0; r < 8; ++r)
        sS[(r + 8 * hi) * 32 + sub * 16 + m] = s[r] * scale;
    }
    __syncthreads();

    if (lane < 16) {                 // online softmax, one lane per row
      float mold = sMax[lane], mx = mold;
      for (int c = 0; c < 32; ++c) mx = fmaxf(mx, sS[lane * 32 + c]);
      const float al = __expf(mold - mx);
      float ts = 0.f;
      for (int c = 0; c < 32; ++c) {
        const float p = __expf(sS[lane * 32 + c] - mx);
        ts += p;
        sP[lane * 32 + c] = (_Float16)p;
      }
      sSum[lane] = sSum[lane] * al + ts;
      sMax[lane] = mx;
      sAlpha[lane] = al;
    }
    __syncthreads();

#pragma unroll
    for (int r = 0; r < 8; ++r) {    // rescale O accumulators
      const float al = sAlpha[r + 8 * hi];
      o0[r] *= al; o1[r] *= al; o2[r] *= al;
    }
    v16h ap;                          // P tile as A fragment (ds_load_b128 x2)
    {
      const v8h p0 = *(const v8h*)&sP[m * 32 + hi * 8];
      const v8h p1 = *(const v8h*)&sP[m * 32 + 16 + hi * 8];
#pragma unroll
      for (int e = 0; e < 8; ++e) { ap[e] = p0[e]; ap[e + 8] = p1[e]; }
    }
    v16h bv0, bv1, bv2;               // V as B fragments, 3 col tiles
#pragma unroll
    for (int e = 0; e < 16; ++e) {
      const float* vp = V + (long)(kt + (e & 8) * 2 + hi * 8 + (e & 7)) * MHA_D + h * MHA_HD;
      bv0[e] = (_Float16)vp[m];
      bv1[e] = (_Float16)vp[16 + m];
      bv2[e] = (_Float16)vp[32 + m];
    }
    o0 = wmma_f16(ap, bv0, o0);
    o1 = wmma_f16(ap, bv1, o1);
    o2 = wmma_f16(ap, bv2, o2);
    __syncthreads();
  }

#pragma unroll
  for (int r = 0; r < 8; ++r) {
    const int row = q0 + r + 8 * hi;
    const float inv = 1.f / sSum[r + 8 * hi];
    float* op = O + (long)row * MHA_D + h * MHA_HD;
    op[m]      = o0[r] * inv;
    op[16 + m] = o1[r] * inv;
    op[32 + m] = o2[r] * inv;
  }
}

// ---------------------------------------------------------------------------
// gene embedding gather + L2 normalize, writes x_in cols 192..255
// ---------------------------------------------------------------------------
__global__ void k_gene(const float* __restrict__ table,
                       const int* __restrict__ gidx,
                       float* __restrict__ x_in)
{
  const int n = blockIdx.x, j = threadIdx.x;  // 64 threads
  int g = gidx[n];
  g = g < 0 ? 0 : (g > 19999 ? 19999 : g);
  const float v = table[(long)g * 64 + j];
  __shared__ float sh[64];
  sh[j] = v * v;
  __syncthreads();
  for (int s = 32; s > 0; s >>= 1) { if (j < s) sh[j] += sh[j + s]; __syncthreads(); }
  const float norm = fmaxf(sqrtf(sh[0]), 1e-12f);
  x_in[(long)n * HIDC + MHA_D + j] = v / norm;
}

__global__ void k_fill(float* p, float v, long n) {
  const long i = (long)blockIdx.x * 256 + threadIdx.x;
  if (i < n) p[i] = v;
}

__global__ void k_copy(const float* __restrict__ s, float* __restrict__ d, int n) {
  const int i = blockIdx.x * 256 + threadIdx.x;
  if (i < n) d[i] = s[i];
}

__device__ inline int pid_of(int a) { return a < 0 ? 15 : (a > 15 ? 15 : a); }

// alpha[e,h] = dot(q[dst,h,:], k[src,h,:] + eproj[pid,h,:]) / 16 ; amax per dst
__global__ void k_edge_alpha(const float* __restrict__ qn,
                             const float* __restrict__ kn,
                             const float* __restrict__ tproj,
                             const int* __restrict__ src,
                             const int* __restrict__ dst,
                             const int* __restrict__ attr,
                             float* __restrict__ alpha,
                             unsigned* __restrict__ amax, int H)
{
  const int e = blockIdx.x;
  const int h = threadIdx.x >> 5;
  const int lane = threadIdx.x & 31;
  const int s = src[e], d = dst[e], p = pid_of(attr[e]);
  const long HC = (long)H * HIDC;
  const float* qp = qn + (long)d * HC + h * HIDC;
  const float* kp = kn + (long)s * HC + h * HIDC;
  const float* ep = tproj + (long)p * HC + h * HIDC;
  float acc = 0.f;
#pragma unroll
  for (int c0 = 0; c0 < HIDC; c0 += 128) {
    const v4f qv = *(const v4f*)(qp + c0 + lane * 4);
    const v4f kv = *(const v4f*)(kp + c0 + lane * 4);
    const v4f ev = *(const v4f*)(ep + c0 + lane * 4);
#pragma unroll
    for (int i = 0; i < 4; ++i) acc += qv[i] * (kv[i] + ev[i]);
  }
  for (int off = 16; off > 0; off >>= 1) acc += __shfl_xor(acc, off, 32);
  if (lane == 0) {
    const float a = acc * (1.f / 16.f);          // / sqrt(256)
    alpha[(long)e * H + h] = a;
    atomicMax(&amax[(long)d * H + h], fkey(a));
  }
}

__global__ void k_edge_exp(const float* __restrict__ alpha,
                           const unsigned* __restrict__ amax,
                           const int* __restrict__ dst,
                           float* __restrict__ ex, float* __restrict__ den,
                           int H)
{
  const int i = blockIdx.x * 256 + threadIdx.x;
  if (i >= N_EDGES * H) return;
  const int e = i / H, h = i - e * H;
  const int d = dst[e];
  const float v = __expf(alpha[i] - funkey(amax[(long)d * H + h]));
  ex[i] = v;
  atomicAdd(&den[(long)d * H + h], v);
}

__global__ void k_edge_scatter(const float* __restrict__ vn,
                               const float* __restrict__ tproj,
                               const float* __restrict__ ex,
                               const float* __restrict__ den,
                               const int* __restrict__ src,
                               const int* __restrict__ dst,
                               const int* __restrict__ attr,
                               float* __restrict__ seg, int H)
{
  const int e = blockIdx.x, c = threadIdx.x;     // 256 threads
  const int s = src[e], d = dst[e], p = pid_of(attr[e]);
  const long HC = (long)H * HIDC;
  for (int h = 0; h < H; ++h) {
    const float w = ex[(long)e * H + h] / (den[(long)d * H + h] + 1e-16f);
    const float val = w * (vn[(long)s * HC + h * HIDC + c] +
                           tproj[(long)p * HC + h * HIDC + c]);
    atomicAdd(&seg[(long)d * HC + h * HIDC + c], val);
  }
}

// (mean over heads if H==4) + skip, LayerNorm, exact GELU
__global__ void k_post_conv(const float* __restrict__ seg,
                            const float* __restrict__ skip,
                            const float* __restrict__ g,
                            const float* __restrict__ b,
                            float* __restrict__ x1, int H)
{
  const int n = blockIdx.x, c = threadIdx.x;     // 256 threads
  float v;
  if (H == 4) {
    const float* s = seg + (long)n * 1024;
    v = 0.25f * (s[c] + s[256 + c] + s[512 + c] + s[768 + c]);
  } else {
    v = seg[(long)n * HIDC + c];
  }
  v += skip[(long)n * HIDC + c];
  __shared__ float sh[HIDC];
  sh[c] = v; __syncthreads();
  for (int s2 = 128; s2 > 0; s2 >>= 1) { if (c < s2) sh[c] += sh[c + s2]; __syncthreads(); }
  const float mean = sh[0] * (1.f / HIDC);
  __syncthreads();
  const float dv = v - mean;
  sh[c] = dv * dv; __syncthreads();
  for (int s2 = 128; s2 > 0; s2 >>= 1) { if (c < s2) sh[c] += sh[c + s2]; __syncthreads(); }
  const float var = sh[0] * (1.f / HIDC);
  const float y = dv * rsqrtf(var + 1e-5f) * g[c] + b[c];
  x1[(long)n * HIDC + c] = 0.5f * y * (1.f + erff(y * 0.70710678118654752f));
}

// beta = sigmoid([xa,xb] @ Wb + bb); out = beta*xa + (1-beta)*xb
__global__ void k_beta_mix(const float* __restrict__ xa,
                           const float* __restrict__ xb,
                           const float* __restrict__ Wb,
                           const float* __restrict__ bb,
                           float* __restrict__ out)
{
  const int n = blockIdx.x, c = threadIdx.x;     // 256 threads
  const float va = xa[(long)n * HIDC + c], v1 = xb[(long)n * HIDC + c];
  __shared__ float sh[HIDC];
  sh[c] = va * Wb[c] + v1 * Wb[HIDC + c];
  __syncthreads();
  for (int s = 128; s > 0; s >>= 1) { if (c < s) sh[c] += sh[c + s]; __syncthreads(); }
  const float beta = 1.f / (1.f + __expf(-(sh[0] + bb[0])));
  out[(long)n * HIDC + c] = beta * va + (1.f - beta) * v1;
}

// ---------------------------------------------------------------------------
// Host side
// ---------------------------------------------------------------------------
static void gemm(hipStream_t st, const float* A, long lda, long acol,
                 const int* arows, const float* W, int ldw, const float* bias,
                 float* C, int ldc, int ccol, int M, int N, int K)
{
  dim3 g((M + 63) / 64, N / 16);
  k_gemm_wmma<<<g, dim3(128), 0, st>>>(A, lda, acol, arows, W, ldw, bias,
                                       C, ldc, ccol, M, N, K);
}
static void fill0(hipStream_t st, float* p, long n) {
  k_fill<<<dim3((unsigned)((n + 255) / 256)), dim3(256), 0, st>>>(p, 0.f, n);
}

// workspace layout (float offsets)
#define OF_XIN   0L
#define OF_XMID  1048576L
#define OF_S     2097152L
#define OF_DNAC  (OF_S)
#define OF_QM    (OF_S + 786432L)
#define OF_KM    (OF_S + 1572864L)
#define OF_VM    (OF_S + 2359296L)
#define OF_OM    (OF_S + 3145728L)
#define OF_QN    (OF_S)
#define OF_KN    (OF_S + 4194304L)
#define OF_VN    (OF_S + 8388608L)
#define OF_TP    (OF_S + 12582912L)
#define OF_AL    (OF_S + 12599296L)
#define OF_EX    (OF_S + 12861440L)
#define OF_AM    (OF_S + 13123584L)
#define OF_DEN   (OF_S + 13139968L)
#define OF_SEG   (OF_S + 13156352L)
#define OF_SKIP  (OF_S + 17350656L)
#define OF_X1    (OF_S + 18399232L)

static void run_conv(hipStream_t st, float* ws, const float* x, int H,
                     const float* const* cp /* Wq bq Wk bk Wv bv We be Wsk bsk */,
                     const float* lng, const float* lnb,
                     const float* Wbg, const float* bbg,
                     const float* ptable,
                     const int* src, const int* dst, const int* attr,
                     float* out)
{
  const int HC = H * HIDC;
  float* qn = ws + OF_QN;  float* kn = ws + OF_KN;  float* vn = ws + OF_VN;
  float* tp = ws + OF_TP;  float* al = ws + OF_AL;  float* ex = ws + OF_EX;
  unsigned* am = (unsigned*)(ws + OF_AM);
  float* den = ws + OF_DEN; float* seg = ws + OF_SEG;
  float* skp = ws + OF_SKIP; float* x1 = ws + OF_X1;

  fill0(st, (float*)am, (long)N_NODES * H);      // key 0 == -inf
  fill0(st, den, (long)N_NODES * H);
  fill0(st, seg, (long)N_NODES * HC);

  gemm(st, x, HIDC, 0, nullptr, cp[0], HC, cp[1], qn, HC, 0, N_NODES, HC, HIDC);
  gemm(st, x, HIDC, 0, nullptr, cp[2], HC, cp[3], kn, HC, 0, N_NODES, HC, HIDC);
  gemm(st, x, HIDC, 0, nullptr, cp[4], HC, cp[5], vn, HC, 0, N_NODES, HC, HIDC);
  gemm(st, ptable, 32, 0, nullptr, cp[6], HC, cp[7], tp, HC, 0, 16, HC, 32);
  gemm(st, x, HIDC, 0, nullptr, cp[8], HIDC, cp[9], skp, HIDC, 0,
       N_NODES, HIDC, HIDC);

  k_edge_alpha<<<dim3(N_EDGES), dim3(32 * H), 0, st>>>(qn, kn, tp, src, dst,
                                                       attr, al, am, H);
  k_edge_exp<<<dim3((N_EDGES * H + 255) / 256), dim3(256), 0, st>>>(al, am, dst,
                                                                    ex, den, H);
  k_edge_scatter<<<dim3(N_EDGES), dim3(256), 0, st>>>(vn, tp, ex, den, src, dst,
                                                      attr, seg, H);
  k_post_conv<<<dim3(N_NODES), dim3(256), 0, st>>>(seg, skp, lng, lnb, x1, H);
  k_beta_mix<<<dim3(N_NODES), dim3(256), 0, st>>>(x, x1, Wbg, bbg, out);
}

extern "C" void kernel_launch(void* const* d_in, const int* in_sizes, int n_in,
                              void* d_out, int out_size, void* d_ws, size_t ws_size,
                              hipStream_t stream)
{
  // params flattened in declaration (insertion) order, then the other inputs
  const float* gene_table = (const float*)d_in[0];
  const float* ptable     = (const float*)d_in[1];
  const float* W1 = (const float*)d_in[2];  const float* b1 = (const float*)d_in[3];
  const float* W2 = (const float*)d_in[4];  const float* b2 = (const float*)d_in[5];
  const float* W3 = (const float*)d_in[6];  const float* b3 = (const float*)d_in[7];
  const float* mWq = (const float*)d_in[8];  const float* mbq = (const float*)d_in[9];
  const float* mWk = (const float*)d_in[10]; const float* mbk = (const float*)d_in[11];
  const float* mWv = (const float*)d_in[12]; const float* mbv = (const float*)d_in[13];
  const float* mWo = (const float*)d_in[14]; const float* mbo = (const float*)d_in[15];
  const float* conv1[10]; for (int i = 0; i < 10; ++i) conv1[i] = (const float*)d_in[16 + i];
  const float* conv2[10]; for (int i = 0; i < 10; ++i) conv2[i] = (const float*)d_in[26 + i];
  const float* ln1g = (const float*)d_in[36]; const float* ln1b = (const float*)d_in[37];
  const float* ln2g = (const float*)d_in[38]; const float* ln2b = (const float*)d_in[39];
  const float* Wb1 = (const float*)d_in[40];  const float* bb1 = (const float*)d_in[41];
  const float* Wb2 = (const float*)d_in[42];  const float* bb2 = (const float*)d_in[43];
  const float* combined = (const float*)d_in[44];
  const int* gidx = (const int*)d_in[45];
  const int* didx = (const int*)d_in[46];
  const int* eidx = (const int*)d_in[47];
  const int* attr = (const int*)d_in[48];
  const int* src = eidx;
  const int* dst = eidx + N_EDGES;

  float* ws  = (float*)d_ws;
  float* out = (float*)d_out;
  float* x_in  = ws + OF_XIN;
  float* x_mid = ws + OF_XMID;
  float* dnac  = ws + OF_DNAC;
  float* qm = ws + OF_QM; float* km = ws + OF_KM; float* vm = ws + OF_VM;
  float* om = ws + OF_OM;

  // gene embeddings -> x_in[:,192:256]
  k_gene<<<dim3(N_NODES), dim3(64), 0, stream>>>(gene_table, gidx, x_in);

  // input projections -> dna_cat [4096,192]
  gemm(stream, combined, 4608, 0,    didx, W1, 64, b1, dnac, MHA_D, 0,   N_NODES, 64, 1024);
  gemm(stream, combined, 4608, 1024, didx, W2, 64, b2, dnac, MHA_D, 64,  N_NODES, 64, 1024);
  gemm(stream, combined, 4608, 2048, didx, W3, 64, b3, dnac, MHA_D, 128, N_NODES, 64, 2560);

  // MHA qkv, attention, output proj -> x_in[:,0:192]
  gemm(stream, dnac, MHA_D, 0, nullptr, mWq, MHA_D, mbq, qm, MHA_D, 0, N_NODES, MHA_D, MHA_D);
  gemm(stream, dnac, MHA_D, 0, nullptr, mWk, MHA_D, mbk, km, MHA_D, 0, N_NODES, MHA_D, MHA_D);
  gemm(stream, dnac, MHA_D, 0, nullptr, mWv, MHA_D, mbv, vm, MHA_D, 0, N_NODES, MHA_D, MHA_D);
  k_mha_attn<<<dim3(N_NODES / 16, 4), dim3(32), 0, stream>>>(qm, km, vm, om);
  gemm(stream, om, MHA_D, 0, nullptr, mWo, MHA_D, mbo, x_in, HIDC, 0, N_NODES, MHA_D, MHA_D);

  // conv1 (4 heads, mean) -> x_mid ; conv2 (1 head) -> d_out
  run_conv(stream, ws, x_in,  4, conv1, ln1g, ln1b, Wb1, bb1, ptable, src, dst, attr, x_mid);
  run_conv(stream, ws, x_mid, 1, conv2, ln2g, ln2b, Wb2, bb2, ptable, src, dst, attr, out);

  // second tuple output: pathway_table passthrough
  k_copy<<<dim3(2), dim3(256), 0, stream>>>(ptable, out + (long)N_NODES * HIDC, 512);
}
